// SelfReferentialTitan_2439541424811
// MI455X (gfx1250) — compile-verified, matrix operand order checked by reading, not run on hardware
//
#include <hip/hip_runtime.h>
#include <math.h>

// ---------------------------------------------------------------------------
// SelfReferentialTitan fused kernel for gfx1250 (MI455X)
//   x:[N,F] f32, W1:[6,F,H], B1:[6,H], W2:[6,H,F], B2:[6,F]
//   out:[6,N,F] f32 = [k, v, q, softplus(y3), sigmoid(y4), x + mmemory(q)]
// bf16 WMMA (16x16x32) + f32 accum; full fusion: HBM = read x + write 6 outs.
// M_TILE=64 + register-level B-fragment reuse to cut L2 weight traffic ~2.6x.
// ---------------------------------------------------------------------------

#define N_TOK   32768
#define F_DIM   1024
#define H_DIM   64
#define NBR     6
#define M_TILE  64            // rows per block
#define LDA_X   1032          // padded bf16 row stride for x/q LDS tiles
#define LDA_H   72            // padded bf16 row stride for h LDS tile

typedef __attribute__((ext_vector_type(16))) __bf16 v16bf;
typedef __attribute__((ext_vector_type(8)))  float  v8f;

union Frag16 { v16bf bf; int4 i4[2]; };

// Load a 16x32 bf16 WMMA operand fragment from a [row][K] row-major array.
// Per ISA 7.12.2 (16-bit A layout): lane = row (mod 16); lanes 0-15 hold
// K = [kb..kb+8) and [kb+16..kb+24); lanes 16-31 the other K halves.
// Each half is one contiguous 16-byte chunk -> two b128 loads.
static __device__ __forceinline__ v16bf load_frag(const __bf16* base, int row,
                                                  int stride, int kbase, int lhalf) {
  Frag16 f;
  const __bf16* p = base + row * stride + kbase + lhalf * 8;
  f.i4[0] = *reinterpret_cast<const int4*>(p);
  f.i4[1] = *reinterpret_cast<const int4*>(p + 16);
  return f.bf;
}

static __device__ __forceinline__ v8f wmma_bf16(v16bf a, v16bf b, v8f c) {
  return __builtin_amdgcn_wmma_f32_16x16x32_bf16(false, a, false, b,
                                                 (short)0, c, false, false);
}

// ---------------------------------------------------------------------------
// Pre-pass: convert weights to bf16, transposed to [N][K] row-major so the
// B-matrix fragment loads use the same contiguous-chunk pattern as A.
//   W1t[m][h][f] = bf16(W1[m][f][h])   (6 x 64 x 1024)
//   W2t[m][f][h] = bf16(W2[m][h][f])   (6 x 1024 x 64)
// ---------------------------------------------------------------------------
__global__ void convert_weights(const float* __restrict__ W1,
                                const float* __restrict__ W2,
                                unsigned short* __restrict__ w1t_u,
                                unsigned short* __restrict__ w2t_u) {
  __bf16* W1t = reinterpret_cast<__bf16*>(w1t_u);
  __bf16* W2t = reinterpret_cast<__bf16*>(w2t_u);
  const int idx = blockIdx.x * 256 + threadIdx.x;   // 6*1024*64 = 393216 total
  if (idx >= NBR * F_DIM * H_DIM) return;
  const int m   = idx >> 16;          // /65536
  const int rem = idx & 65535;
  const int f1 = rem >> 6, h1 = rem & 63;            // W1: [m][f][h]
  W1t[(m << 16) + h1 * F_DIM + f1] = (__bf16)W1[idx];
  const int h2 = rem >> 10, f2 = rem & 1023;         // W2: [m][h][f]
  W2t[(m << 16) + f2 * H_DIM + h2] = (__bf16)W2[idx];
}

// ---------------------------------------------------------------------------
// Fused main kernel: one block = 64 rows, 8 waves, all 6 branches.
// ---------------------------------------------------------------------------
__global__ __launch_bounds__(256, 1)
void titan_fused(const float* __restrict__ x,
                 const float* __restrict__ B1,
                 const float* __restrict__ B2,
                 const unsigned short* __restrict__ w1t_u,
                 const unsigned short* __restrict__ w2t_u,
                 float* __restrict__ out) {
  extern __shared__ char smem[];
  __bf16* x_lds = reinterpret_cast<__bf16*>(smem);        // [64][1032]
  __bf16* q_lds = x_lds + M_TILE * LDA_X;                 // [64][1032]
  __bf16* h_lds = q_lds + M_TILE * LDA_X;                 // [64][72]
  const __bf16* W1t = reinterpret_cast<const __bf16*>(w1t_u);
  const __bf16* W2t = reinterpret_cast<const __bf16*>(w2t_u);

  const int tid   = threadIdx.x;
  const int wave  = tid >> 5;
  const int lane  = tid & 31;
  const int l16   = lane & 15;
  const int lhalf = lane >> 4;
  const int row0  = blockIdx.x * M_TILE;
  const size_t NF = (size_t)N_TOK * F_DIM;

  // ---- stage x tile in LDS as bf16 (coalesced b128 reads, b64 LDS writes)
  {
    const int col = tid * 4;                 // 256 threads x 4 floats = 1024
    for (int r = 0; r < M_TILE; ++r) {
      const float4 v =
          *reinterpret_cast<const float4*>(x + (size_t)(row0 + r) * F_DIM + col);
      union { __bf16 h[4]; uint2 u; } pk;
      pk.h[0] = (__bf16)v.x; pk.h[1] = (__bf16)v.y;
      pk.h[2] = (__bf16)v.z; pk.h[3] = (__bf16)v.w;
      *reinterpret_cast<uint2*>(x_lds + r * LDA_X + col) = pk.u;
    }
  }
  __syncthreads();

  for (int b = 0; b < NBR; ++b) {
    const __bf16* aSrc = (b == 5) ? q_lds : x_lds;     // mmemory takes q
    const __bf16* w1   = W1t + b * (H_DIM * F_DIM);
    const __bf16* w2   = W2t + b * (H_DIM * F_DIM);
    __builtin_prefetch(w2, 0, 0);                      // global_prefetch_b8

    // ---- fc1: h = relu(a @ W1 + b1)
    // Each wave: one nsub, two msub accumulators -> B-frag reused 2x.
    {
      const int nsub  = wave & 3;        // H = 4 col tiles
      const int mbase = (wave >> 2) * 2; // rows: waves 0-3 -> 0,1; 4-7 -> 2,3
      v8f c0 = {}, c1 = {};
#pragma unroll 4
      for (int ks = 0; ks < F_DIM / 32; ++ks) {
        v16bf bw = load_frag(w1, nsub * 16 + l16, F_DIM, ks * 32, lhalf);
        v16bf a0 = load_frag(aSrc, (mbase + 0) * 16 + l16, LDA_X, ks * 32, lhalf);
        v16bf a1 = load_frag(aSrc, (mbase + 1) * 16 + l16, LDA_X, ks * 32, lhalf);
        c0 = wmma_bf16(a0, bw, c0);
        c1 = wmma_bf16(a1, bw, c1);
      }
      const int   n   = nsub * 16 + l16;
      const float b1v = B1[b * H_DIM + n];
#pragma unroll
      for (int r = 0; r < 8; ++r) {
        const int mr = r + lhalf * 8;                  // C layout: M=r(+8)
        float h0 = fmaxf(c0[r] + b1v, 0.0f);
        float h1 = fmaxf(c1[r] + b1v, 0.0f);
        h_lds[((mbase + 0) * 16 + mr) * LDA_H + n] = (__bf16)h0;
        h_lds[((mbase + 1) * 16 + mr) * LDA_H + n] = (__bf16)h1;
      }
    }
    __syncthreads();

    // ---- fc2 + epilogue: y = a + h @ W2 + b2 (+activation)
    // Each wave: 8 nsubs; per nsub all 4 msubs share one B-frag (reused 4x).
    for (int nt = 0; nt < 8; ++nt) {
      const int nsub = wave * 8 + nt;    // 64 col tiles of F
      v8f acc[4] = {{}, {}, {}, {}};
#pragma unroll
      for (int ks = 0; ks < 2; ++ks) {   // K = 64
        v16bf bw = load_frag(w2, nsub * 16 + l16, H_DIM, ks * 32, lhalf);
#pragma unroll
        for (int ms = 0; ms < 4; ++ms) {
          v16bf a = load_frag(h_lds, ms * 16 + l16, LDA_H, ks * 32, lhalf);
          acc[ms] = wmma_bf16(a, bw, acc[ms]);
        }
      }
      const int   n   = nsub * 16 + l16;
      const float b2v = B2[b * F_DIM + n];
#pragma unroll
      for (int ms = 0; ms < 4; ++ms) {
#pragma unroll
        for (int r = 0; r < 8; ++r) {
          const int m = ms * 16 + r + lhalf * 8;
          float res = (float)x_lds[m * LDA_X + n];
          if (b == 5) res += (float)q_lds[m * LDA_X + n]; // out = x + (q + mlp)
          const float yv = acc[ms][r] + b2v + res;
          float o;
          if (b == 3)      o = (yv > 20.0f) ? yv : log1pf(__expf(yv)); // softplus
          else if (b == 4) o = 1.0f / (1.0f + __expf(-yv));            // sigmoid
          else             o = yv;                                     // k,v,q,out
          if (b == 2) q_lds[m * LDA_X + n] = (__bf16)yv;               // keep q
          out[(size_t)b * NF + (size_t)(row0 + m) * F_DIM + n] = o;
        }
      }
    }
    __syncthreads();   // h_lds reused next branch; q_lds complete before b==5
  }
}

// ---------------------------------------------------------------------------
extern "C" void kernel_launch(void* const* d_in, const int* in_sizes, int n_in,
                              void* d_out, int out_size, void* d_ws, size_t ws_size,
                              hipStream_t stream) {
  const float* x  = (const float*)d_in[0];
  const float* W1 = (const float*)d_in[1];
  const float* B1 = (const float*)d_in[2];
  const float* W2 = (const float*)d_in[3];
  const float* B2 = (const float*)d_in[4];

  unsigned short* W1t = (unsigned short*)d_ws;                  // 6*64*1024 bf16
  unsigned short* W2t = W1t + NBR * H_DIM * F_DIM;              // 6*1024*64 bf16

  const int welems = NBR * F_DIM * H_DIM;                       // 393216
  convert_weights<<<(welems + 255) / 256, 256, 0, stream>>>(W1, W2, W1t, W2t);

  const size_t smem_bytes =
      (size_t)(2 * M_TILE * LDA_X + M_TILE * LDA_H) * sizeof(unsigned short); // 273408
  titan_fused<<<N_TOK / M_TILE, 256, smem_bytes, stream>>>(
      x, B1, B2, W1t, W2t, (float*)d_out);
}
